// Decoder_9698036155160
// MI455X (gfx1250) — matrix operand order repaired and checked by reading, not compiled, not run on hardware
//
#include <hip/hip_runtime.h>

// GRU decoder, persistent-workgroup formulation for MI455X (gfx1250, wave32).
// Each workgroup owns BT=16 batch rows and carries the recurrence through all
// 127 steps; every GEMM runs on v_wmma_f32_16x16x32_bf16 with fp32 accum.
// Activations use v_exp_f32 + v_rcp_f32 (no IEEE-division expansion).

#define LATENT_D 256
#define H_D      512
#define A_D      64
#define L_D      128
#define BT       16
#define NWAVE    8
#define NTHR     256

typedef __bf16 bf16_t;
typedef bf16_t bf16x8  __attribute__((ext_vector_type(8)));
typedef bf16_t bf16x16 __attribute__((ext_vector_type(16)));
typedef float  f32x8   __attribute__((ext_vector_type(8)));

// A-operand fragment (16x32 bf16): lane holds row (lane&15); K elements are the
// two interleaved 8-element runs  [k0 + lh*8, +8)  and  [k0+16 + lh*8, +8).
__device__ __forceinline__ bf16x16 ldfragA(const bf16_t* row, int k0, int lh) {
  bf16x8 lo = *(const bf16x8*)(row + k0 + lh * 8);
  bf16x8 hi = *(const bf16x8*)(row + k0 + 16 + lh * 8);
  return __builtin_shufflevector(lo, hi, 0, 1, 2, 3, 4, 5, 6, 7,
                                 8, 9, 10, 11, 12, 13, 14, 15);
}

// B-operand fragment (32x16 bf16, fed as N x K rows of W): lane holds output
// column (lane&15); K elements are the two interleaved 8-element runs
// [k0 + lh*8, +8)  and  [k0+16 + lh*8, +8).
__device__ __forceinline__ bf16x16 ldfragB(const bf16_t* row, int k0, int lh) {
  const bf16_t* p = row + k0 + lh * 8;
  bf16x8 lo = *(const bf16x8*)(p);
  bf16x8 hi = *(const bf16x8*)(p + 16);
  return __builtin_shufflevector(lo, hi, 0, 1, 2, 3, 4, 5, 6, 7,
                                 8, 9, 10, 11, 12, 13, 14, 15);
}

// One 16x16 output tile of  X[16xK] @ W[NxK]^T  accumulated over K.
__device__ __forceinline__ f32x8 gemm_tile(const bf16_t* Abase, int Astride,
                                           const bf16_t* Wn0, int K,
                                           f32x8 acc, int lm, int lh) {
  const bf16_t* arow = Abase + lm * Astride;
  const bf16_t* brow = Wn0 + (size_t)lm * K;
#pragma unroll 4
  for (int k0 = 0; k0 < K; k0 += 32) {
    bf16x16 fa = ldfragA(arow, k0, lh);
    bf16x16 fb = ldfragB(brow, k0, lh);
    acc = __builtin_amdgcn_wmma_f32_16x16x32_bf16(false, fa, false, fb,
                                                  (short)0, acc, false, false);
  }
  return acc;
}

// Fast activations: one v_exp_f32 + one v_rcp_f32 each (avoid div expansion).
__device__ __forceinline__ float sigm_f(float x) {
  return __builtin_amdgcn_rcpf(1.0f + __expf(-x));
}
__device__ __forceinline__ float tanh_f(float x) {
  return __builtin_fmaf(2.0f, __builtin_amdgcn_rcpf(1.0f + __expf(-2.0f * x)),
                        -1.0f);
}

__global__ void cvt_f32_bf16(const float* __restrict__ src,
                             bf16_t* __restrict__ dst, int n) {
  int i = blockIdx.x * blockDim.x + threadIdx.x;
  if (i < n) dst[i] = (bf16_t)src[i];
}

__global__ __launch_bounds__(NTHR) void decoder_persistent(
    const float* __restrict__ latent, const float* __restrict__ target,
    const bf16_t* __restrict__ Wd1, const float* __restrict__ bd1,
    const bf16_t* __restrict__ Wd2, const float* __restrict__ bd2,
    const bf16_t* __restrict__ Wd3, const float* __restrict__ bd3,
    const bf16_t* __restrict__ Wih, const bf16_t* __restrict__ Whh,
    const float* __restrict__ b_ih, const float* __restrict__ b_hh,
    const bf16_t* __restrict__ Wm1, const float* __restrict__ bm1,
    const bf16_t* __restrict__ Wm2, const float* __restrict__ bm2,
    const bf16_t* __restrict__ Wm3, const float* __restrict__ bm3,
    float* __restrict__ out) {
  constexpr int HS = H_D + 8;      // padded LDS row stride (1040 B, 16B-aligned)
  constexpr int XS = A_D + 8;      // 144 B
  constexpr int GS = 4 * H_D + 8;  // gate buffer stride

  __shared__ alignas(16) bf16_t hS[BT][HS];  // hidden state (bf16)
  __shared__ alignas(16) bf16_t xS[BT][XS];  // step input
  __shared__ alignas(16) bf16_t pS[BT][HS];  // latent / t2 / p1 staging
  __shared__ alignas(16) bf16_t qS[BT][XS];  // p2 staging
  __shared__ alignas(16) bf16_t gS[BT][GS];  // [r|z|h_n|i_n] pre-activations

  const int tid = threadIdx.x;
  const int wave = tid >> 5;
  const int lane = tid & 31;
  const int lm = lane & 15;
  const int lh = lane >> 4;
  const int b0 = blockIdx.x * BT;

  // ---- stage latent tile (bf16) into pS; emit y[:,0,:] = bos ----
  for (int e = tid; e < BT * LATENT_D; e += NTHR) {
    int b = e >> 8, c = e & (LATENT_D - 1);
    pS[b][c] = (bf16_t)latent[(size_t)(b0 + b) * LATENT_D + c];
  }
  for (int e = tid; e < BT * A_D; e += NTHR) {
    int b = e >> 6, a = e & (A_D - 1);
    out[((size_t)(b0 + b) * L_D + 0) * A_D + a] = (a == 0) ? 16.0f : -16.0f;
  }
  __syncthreads();

  // ---- t1 = tanh(lat @ Wd1^T + bd1) -> hS ----
  for (int t = wave; t < H_D / 16; t += NWAVE) {
    int n0 = t * 16;
    f32x8 acc = {};
    acc = gemm_tile(&pS[0][0], HS, Wd1 + (size_t)n0 * LATENT_D, LATENT_D, acc, lm, lh);
    float bb = bd1[n0 + lm];
#pragma unroll
    for (int r = 0; r < 8; ++r)
      hS[lh * 8 + r][n0 + lm] = (bf16_t)tanh_f(acc[r] + bb);
  }
  __syncthreads();

  // ---- t2 = tanh(t1 @ Wd2^T + bd2) -> pS ----
  for (int t = wave; t < H_D / 16; t += NWAVE) {
    int n0 = t * 16;
    f32x8 acc = {};
    acc = gemm_tile(&hS[0][0], HS, Wd2 + (size_t)n0 * H_D, H_D, acc, lm, lh);
    float bb = bd2[n0 + lm];
#pragma unroll
    for (int r = 0; r < 8; ++r)
      pS[lh * 8 + r][n0 + lm] = (bf16_t)tanh_f(acc[r] + bb);
  }
  __syncthreads();

  // ---- h0 = t2 @ Wd3^T + bd3 -> hS ----
  for (int t = wave; t < H_D / 16; t += NWAVE) {
    int n0 = t * 16;
    f32x8 acc = {};
    acc = gemm_tile(&pS[0][0], HS, Wd3 + (size_t)n0 * H_D, H_D, acc, lm, lh);
    float bb = bd3[n0 + lm];
#pragma unroll
    for (int r = 0; r < 8; ++r)
      hS[lh * 8 + r][n0 + lm] = (bf16_t)(acc[r] + bb);
  }
  __syncthreads();

  // ---- recurrence: 127 teacher-forced steps ----
  for (int j = 0; j < L_D - 1; ++j) {
    // input tile: bos at step 0, else target[:, j, :]
    for (int e = tid; e < BT * A_D; e += NTHR) {
      int b = e >> 6, a = e & (A_D - 1);
      float v = (j == 0) ? ((a == 0) ? 16.0f : -16.0f)
                         : target[((size_t)(b0 + b) * L_D + j) * A_D + a];
      xS[b][a] = (bf16_t)v;
    }
    __syncthreads();

    // gate GEMMs. jobs 0..63: fused r/z (x@Wih^T + h@Whh^T); 64..95: h_n part;
    // 96..127: i_n part (kept separate for n = tanh(i_n + r*h_n)).
    for (int t = wave; t < 128; t += NWAVE) {
      f32x8 acc = {};
      if (t < 64) {
        int n0 = t * 16;
        acc = gemm_tile(&hS[0][0], HS, Whh + (size_t)n0 * H_D, H_D, acc, lm, lh);
        acc = gemm_tile(&xS[0][0], XS, Wih + (size_t)n0 * A_D, A_D, acc, lm, lh);
        float bb = b_ih[n0 + lm] + b_hh[n0 + lm];
#pragma unroll
        for (int r = 0; r < 8; ++r)
          gS[lh * 8 + r][n0 + lm] = (bf16_t)(acc[r] + bb);
      } else if (t < 96) {
        int n0 = 1024 + (t - 64) * 16;
        acc = gemm_tile(&hS[0][0], HS, Whh + (size_t)n0 * H_D, H_D, acc, lm, lh);
        float bb = b_hh[n0 + lm];
#pragma unroll
        for (int r = 0; r < 8; ++r)
          gS[lh * 8 + r][n0 + lm] = (bf16_t)(acc[r] + bb);
      } else {
        int nc = (t - 96) * 16;
        acc = gemm_tile(&xS[0][0], XS, Wih + (size_t)(1024 + nc) * A_D, A_D, acc, lm, lh);
        float bb = b_ih[1024 + nc + lm];
#pragma unroll
        for (int r = 0; r < 8; ++r)
          gS[lh * 8 + r][1536 + nc + lm] = (bf16_t)(acc[r] + bb);
      }
    }
    __syncthreads();

    // elementwise GRU update, h kept in bf16
    for (int e = tid; e < BT * H_D; e += NTHR) {
      int b = e >> 9, c = e & (H_D - 1);
      float rr = sigm_f((float)gS[b][c]);
      float zz = sigm_f((float)gS[b][512 + c]);
      float hn = (float)gS[b][1024 + c];
      float iv = (float)gS[b][1536 + c];
      float nn = tanh_f(__builtin_fmaf(rr, hn, iv));
      float ho = (float)hS[b][c];
      hS[b][c] = (bf16_t)((1.0f - zz) * nn + zz * ho);
    }
    __syncthreads();

    // p1 = tanh(h_new @ Wm1^T + bm1) -> pS
    for (int t = wave; t < H_D / 16; t += NWAVE) {
      int n0 = t * 16;
      f32x8 acc = {};
      acc = gemm_tile(&hS[0][0], HS, Wm1 + (size_t)n0 * H_D, H_D, acc, lm, lh);
      float bb = bm1[n0 + lm];
#pragma unroll
      for (int r = 0; r < 8; ++r)
        pS[lh * 8 + r][n0 + lm] = (bf16_t)tanh_f(acc[r] + bb);
    }
    __syncthreads();

    // p2 = tanh(p1 @ Wm2^T + bm2) -> qS
    for (int t = wave; t < A_D / 16; t += NWAVE) {
      int n0 = t * 16;
      f32x8 acc = {};
      acc = gemm_tile(&pS[0][0], HS, Wm2 + (size_t)n0 * H_D, H_D, acc, lm, lh);
      float bb = bm2[n0 + lm];
#pragma unroll
      for (int r = 0; r < 8; ++r)
        qS[lh * 8 + r][n0 + lm] = (bf16_t)tanh_f(acc[r] + bb);
    }
    __syncthreads();

    // pred = p2 @ Wm3^T + bm3 -> out[:, j+1, :]  (fp32)
    for (int t = wave; t < A_D / 16; t += NWAVE) {
      int n0 = t * 16;
      f32x8 acc = {};
      acc = gemm_tile(&qS[0][0], XS, Wm3 + (size_t)n0 * A_D, A_D, acc, lm, lh);
      float bb = bm3[n0 + lm];
#pragma unroll
      for (int r = 0; r < 8; ++r)
        out[((size_t)(b0 + lh * 8 + r) * L_D + (j + 1)) * A_D + n0 + lm] =
            acc[r] + bb;
    }
  }
}

extern "C" void kernel_launch(void* const* d_in, const int* in_sizes, int n_in,
                              void* d_out, int out_size, void* d_ws, size_t ws_size,
                              hipStream_t stream) {
  (void)n_in; (void)out_size; (void)ws_size;
  const float* latent = (const float*)d_in[0];
  const float* target = (const float*)d_in[1];
  const float* Wd1  = (const float*)d_in[2];
  const float* bd1  = (const float*)d_in[3];
  const float* Wd2  = (const float*)d_in[4];
  const float* bd2  = (const float*)d_in[5];
  const float* Wd3  = (const float*)d_in[6];
  const float* bd3  = (const float*)d_in[7];
  const float* W_ih = (const float*)d_in[8];
  const float* W_hh = (const float*)d_in[9];
  const float* b_ih = (const float*)d_in[10];
  const float* b_hh = (const float*)d_in[11];
  const float* Wm1  = (const float*)d_in[12];
  const float* bm1  = (const float*)d_in[13];
  const float* Wm2  = (const float*)d_in[14];
  const float* bm2  = (const float*)d_in[15];
  const float* Wm3  = (const float*)d_in[16];
  const float* bm3  = (const float*)d_in[17];
  float* out = (float*)d_out;
  bf16_t* ws = (bf16_t*)d_ws;

  // bf16 weight images in workspace (element offsets, all 16B-aligned)
  const size_t oWd1 = 0;
  const size_t oWd2 = oWd1 + 512 * 256;      // 131072
  const size_t oWd3 = oWd2 + 512 * 512;      // 393216
  const size_t oWih = oWd3 + 512 * 512;      // 655360
  const size_t oWhh = oWih + 1536 * 64;      // 753664
  const size_t oWm1 = oWhh + 1536 * 512;     // 1540096
  const size_t oWm2 = oWm1 + 512 * 512;      // 1802240
  const size_t oWm3 = oWm2 + 64 * 512;       // 1835008

  struct Cv { const float* s; size_t o; int n; } cv[8] = {
    {Wd1, oWd1, 512 * 256}, {Wd2, oWd2, 512 * 512}, {Wd3, oWd3, 512 * 512},
    {W_ih, oWih, 1536 * 64}, {W_hh, oWhh, 1536 * 512},
    {Wm1, oWm1, 512 * 512}, {Wm2, oWm2, 64 * 512}, {Wm3, oWm3, 64 * 64}};
  for (int i = 0; i < 8; ++i)
    cvt_f32_bf16<<<(cv[i].n + 255) / 256, 256, 0, stream>>>(cv[i].s, ws + cv[i].o, cv[i].n);

  int batch = in_sizes[0] / LATENT_D;  // 4096
  decoder_persistent<<<batch / BT, NTHR, 0, stream>>>(
      latent, target,
      ws + oWd1, bd1, ws + oWd2, bd2, ws + oWd3, bd3,
      ws + oWih, ws + oWhh, b_ih, b_hh,
      ws + oWm1, bm1, ws + oWm2, bm2, ws + oWm3, bm3,
      out);
}
